// Memory_pooling_41291815584280
// MI455X (gfx1250) — compile-verified, matrix operand order checked by reading, not run on hardware
//
#include <hip/hip_runtime.h>
#include <hip/hip_bf16.h>

// ---------------------------------------------------------------------------
// Memory_pooling for MI455X (gfx1250).  Every matmul-like stage is mapped to
// v_wmma_f32_16x16x32_f16 (f16 in, f32 accumulate).  wave32; each wave owns a
// 32x32 C tile (2x2 WMMA accumulators), K advanced 32 per step.
//
// WMMA fragment layout assumptions (CDNA5 ISA 7.12.2):
//   A (16-bit, 16x32): lane&15 = M row; element e -> K = (e&7) + (e>>3)*16 + (lane>>4)*8
//   B (16-bit, 32x16): mirror of A with lane&15 = N column, same K map
//   C/D (f32, 16x16):  lane&15 = N; VGPR r -> M = r + (lane>>4)*8
//
// Data layouts are picked so that every A *and* B fragment of the attention
// GEMMs is two contiguous, 16B-aligned 8-half loads:
//   Q matrices      : natural pooled flat = row-major [2304,128]
//   K matrices      : pooled with transposed store -> Km^T row-major [4608,128]
//   V (A@V B-oper)  : natural pooled flat of m_val = V^T row-major [256,4608]
// ---------------------------------------------------------------------------

typedef __attribute__((ext_vector_type(16))) _Float16 v16h;
typedef __attribute__((ext_vector_type(8)))  _Float16 v8h;
typedef __attribute__((ext_vector_type(8)))  float    v8f;

#define DEVINL __device__ __forceinline__

// Problem constants
#define NB   4        // batch
#define KD   128      // KEYDIM
#define VD   256      // VALDIM
#define HW2  2304     // 48*48
#define HW4  4608     // 2*48*48

DEVINL v8f wmma16(v16h a, v16h b, v8f c) {
  return __builtin_amdgcn_wmma_f32_16x16x32_f16(
      /*neg_a=*/false, a, /*neg_b=*/false, b,
      /*c_mod=*/(short)0, c, /*reuse_a=*/false, /*reuse_b=*/false);
}

// element e of a 16-bit A/B fragment maps to this K offset within the 32-step
DEVINL int kmap(int e, int lane) {
  return (e & 7) + ((e >> 3) << 4) + ((lane >> 4) << 3);
}

// A/B fragment load from a K-contiguous f16 matrix row: elements 0..7 and
// 8..15 are two contiguous, 16B-aligned 8-half chunks (ld % 8 == 0).
DEVINL v16h load_row_frag(const _Float16* p) {
  v8h x0 = *reinterpret_cast<const v8h*>(p);
  v8h x1 = *reinterpret_cast<const v8h*>(p + 16);
  v16h r;
#pragma unroll
  for (int e = 0; e < 8; ++e) { r[e] = x0[e]; r[e + 8] = x1[e]; }
  return r;
}

// ---------------------------------------------------------------------------
__global__ __launch_bounds__(256) void k_cvt(const float* __restrict__ s,
                                             _Float16* __restrict__ d, int n) {
  int i = blockIdx.x * blockDim.x + threadIdx.x;
  if (i < n) d[i] = (_Float16)s[i];
}

// ---------------------------------------------------------------------------
// 2x2 stride-2 conv as implicit GEMM: out[co,p] = sum_k W[co,k]*patch[k,p],
// k = ci*4 + kh*2 + kw, p = y*48+x, patch from f32 input (converted on load).
// mode 0: flat        out[b*Cout*2304 + co*2304 + p]          (pooled view)
// mode 1: transposed  out[b*Cout*2304 + ((co&1)*2304+p)*(Cout/2) + co/2]
//                     (the raw-view matrix [Cout/2, 4608], stored transposed)
// mode 2: mem_out hi  out[b*512*2304 + (256+co)*2304 + p]
__global__ __launch_bounds__(128) void k_pool_conv(
    const float* __restrict__ in, const _Float16* __restrict__ w16,
    _Float16* __restrict__ out, int Cin, int Cout, int mode) {
  const int lane = threadIdx.x & 31;
  const int tile = blockIdx.x * 4 + (threadIdx.x >> 5);
  const int TM = Cout >> 5;
  const int tilesPerBatch = TM * 72;
  const int b  = tile / tilesPerBatch;
  const int r  = tile - b * tilesPerBatch;
  const int m0 = (r / 72) * 32;
  const int n0 = (r % 72) * 32;
  const int Ktot = Cin * 4;
  const int kofs = (lane >> 4) << 3;
  const int nlo = n0 + (lane & 15), nhi = nlo + 16;
  const int ylo = nlo / 48, xlo = nlo - ylo * 48;
  const int yhi = nhi / 48, xhi = nhi - yhi * 48;
  const size_t inB = (size_t)b * Cin * 9216;

  v8f acc[2][2] = {};
  for (int kb = 0; kb < Ktot; kb += 32) {
    v16h a0 = load_row_frag(w16 + (size_t)(m0 + (lane & 15)) * Ktot + kb + kofs);
    v16h a1 = load_row_frag(w16 + (size_t)(m0 + 16 + (lane & 15)) * Ktot + kb + kofs);
    v16h b0, b1;
#pragma unroll
    for (int e = 0; e < 16; ++e) {
      int kk = kb + kmap(e, lane);
      int ci = kk >> 2, kh = (kk >> 1) & 1, kw = kk & 1;
      b0[e] = (_Float16)in[inB + ((size_t)ci * 96 + (2 * ylo + kh)) * 96 + (2 * xlo + kw)];
      b1[e] = (_Float16)in[inB + ((size_t)ci * 96 + (2 * yhi + kh)) * 96 + (2 * xhi + kw)];
    }
    acc[0][0] = wmma16(a0, b0, acc[0][0]);
    acc[0][1] = wmma16(a0, b1, acc[0][1]);
    acc[1][0] = wmma16(a1, b0, acc[1][0]);
    acc[1][1] = wmma16(a1, b1, acc[1][1]);
  }
#pragma unroll
  for (int mi = 0; mi < 2; ++mi)
#pragma unroll
    for (int ni = 0; ni < 2; ++ni)
#pragma unroll
      for (int rr = 0; rr < 8; ++rr) {
        int m = m0 + mi * 16 + rr + ((lane >> 4) << 3);
        int n = n0 + ni * 16 + (lane & 15);
        size_t idx;
        if (mode == 0)      idx = ((size_t)b * Cout + m) * HW2 + n;
        else if (mode == 1) idx = (size_t)b * Cout * HW2 +
                                  ((size_t)((m & 1) * HW2 + n)) * (Cout >> 1) + (m >> 1);
        else                idx = (size_t)b * 512 * HW2 + (size_t)(VD + m) * HW2 + n;
        out[idx] = (_Float16)acc[mi][ni][rr];
      }
}

// ---------------------------------------------------------------------------
// Pass 1: per-column (softmax axis = query axis i) max & sum-of-exp of each of
// the 4 logit matrices L_t = Q_t @ K_t  ([2304 x 4608] per batch).
// Q row-major [2304,128]; K supplied as Km^T row-major [4608,128].
__global__ __launch_bounds__(128) void k_qk_stats(
    const _Float16* __restrict__ qk,  const _Float16* __restrict__ pqk,
    const _Float16* __restrict__ mkT, const _Float16* __restrict__ pmkT,
    float* __restrict__ cmax, float* __restrict__ ssum) {
  const int lane = threadIdx.x & 31;
  const int tile = blockIdx.x * 4 + (threadIdx.x >> 5);
  const int mt = tile % 144;
  const int tb = tile / 144;
  const int b  = tb & 3;
  const int t  = tb >> 2;                   // 0:(q,m) 1:(p,p) 2:(p,m) 3:(q,p)
  const _Float16* Q  = ((t == 0 || t == 3) ? qk  : pqk)  + (size_t)b * HW2 * KD;
  const _Float16* Kt = ((t == 0 || t == 2) ? mkT : pmkT) + (size_t)b * HW4 * KD;
  const int n0 = mt * 32;
  const int nlo = n0 + (lane & 15), nhi = nlo + 16;
  const int kofs = (lane >> 4) << 3;

  float runmax[2] = {-1e30f, -1e30f};
  float runsum[2] = {0.f, 0.f};
  for (int i0 = 0; i0 < HW2; i0 += 32) {
    v8f acc[2][2] = {};
#pragma unroll
    for (int kb = 0; kb < KD; kb += 32) {
      v16h a0 = load_row_frag(Q + (size_t)(i0 + (lane & 15)) * KD + kb + kofs);
      v16h a1 = load_row_frag(Q + (size_t)(i0 + 16 + (lane & 15)) * KD + kb + kofs);
      v16h b0 = load_row_frag(Kt + (size_t)nlo * KD + kb + kofs);
      v16h b1 = load_row_frag(Kt + (size_t)nhi * KD + kb + kofs);
      acc[0][0] = wmma16(a0, b0, acc[0][0]);
      acc[0][1] = wmma16(a0, b1, acc[0][1]);
      acc[1][0] = wmma16(a1, b0, acc[1][0]);
      acc[1][1] = wmma16(a1, b1, acc[1][1]);
    }
    // online column softmax update; column n lives in lanes n and n^16
#pragma unroll
    for (int ni = 0; ni < 2; ++ni) {
      float tmax = -1e30f;
#pragma unroll
      for (int mi = 0; mi < 2; ++mi)
#pragma unroll
        for (int rr = 0; rr < 8; ++rr) tmax = fmaxf(tmax, acc[mi][ni][rr]);
      tmax = fmaxf(tmax, __shfl_xor(tmax, 16, 32));
      float nm = fmaxf(runmax[ni], tmax);
      float s = 0.f;
#pragma unroll
      for (int mi = 0; mi < 2; ++mi)
#pragma unroll
        for (int rr = 0; rr < 8; ++rr) s += __expf(acc[mi][ni][rr] - nm);
      s += __shfl_xor(s, 16, 32);
      runsum[ni] = runsum[ni] * __expf(runmax[ni] - nm) + s;
      runmax[ni] = nm;
    }
  }
  if (lane < 16) {
    size_t base = ((size_t)t * NB + b) * HW4;
#pragma unroll
    for (int ni = 0; ni < 2; ++ni) {
      int m = n0 + ni * 16 + lane;
      cmax[base + m] = runmax[ni];
      ssum[base + m] = runsum[ni];
    }
  }
}

// ---------------------------------------------------------------------------
// Pass 2: Asum[i,m] = sum_t exp(L_t[i,m] - cmax_t[m]) / ssum_t[m]  (f16 store)
__global__ __launch_bounds__(128) void k_asum(
    const _Float16* __restrict__ qk,  const _Float16* __restrict__ pqk,
    const _Float16* __restrict__ mkT, const _Float16* __restrict__ pmkT,
    const float* __restrict__ cmax, const float* __restrict__ ssum,
    _Float16* __restrict__ asum) {
  const int lane = threadIdx.x & 31;
  const int tile = blockIdx.x * 4 + (threadIdx.x >> 5);
  const int mt  = tile % 144;
  const int rem = tile / 144;
  const int it  = rem % 72;
  const int b   = rem / 72;
  const int i0 = it * 32, n0 = mt * 32;
  const int nlo = n0 + (lane & 15), nhi = nlo + 16;
  const int kofs = (lane >> 4) << 3;

  v8f psum[2][2] = {};
#pragma unroll
  for (int t = 0; t < 4; ++t) {
    const _Float16* Q  = ((t == 0 || t == 3) ? qk  : pqk)  + (size_t)b * HW2 * KD;
    const _Float16* Kt = ((t == 0 || t == 2) ? mkT : pmkT) + (size_t)b * HW4 * KD;
    v8f acc[2][2] = {};
#pragma unroll
    for (int kb = 0; kb < KD; kb += 32) {
      v16h a0 = load_row_frag(Q + (size_t)(i0 + (lane & 15)) * KD + kb + kofs);
      v16h a1 = load_row_frag(Q + (size_t)(i0 + 16 + (lane & 15)) * KD + kb + kofs);
      v16h b0 = load_row_frag(Kt + (size_t)nlo * KD + kb + kofs);
      v16h b1 = load_row_frag(Kt + (size_t)nhi * KD + kb + kofs);
      acc[0][0] = wmma16(a0, b0, acc[0][0]);
      acc[0][1] = wmma16(a0, b1, acc[0][1]);
      acc[1][0] = wmma16(a1, b0, acc[1][0]);
      acc[1][1] = wmma16(a1, b1, acc[1][1]);
    }
    const float* cb = cmax + ((size_t)t * NB + b) * HW4;
    const float* sb = ssum + ((size_t)t * NB + b) * HW4;
    float c0 = cb[nlo], c1 = cb[nhi];
    float is0 = 1.f / sb[nlo], is1 = 1.f / sb[nhi];
#pragma unroll
    for (int mi = 0; mi < 2; ++mi)
#pragma unroll
      for (int rr = 0; rr < 8; ++rr) {
        psum[mi][0][rr] += __expf(acc[mi][0][rr] - c0) * is0;
        psum[mi][1][rr] += __expf(acc[mi][1][rr] - c1) * is1;
      }
  }
#pragma unroll
  for (int mi = 0; mi < 2; ++mi)
#pragma unroll
    for (int ni = 0; ni < 2; ++ni)
#pragma unroll
      for (int rr = 0; rr < 8; ++rr) {
        int i = i0 + mi * 16 + rr + ((lane >> 4) << 3);
        int m = n0 + ni * 16 + (lane & 15);
        asum[(size_t)b * HW2 * HW4 + (size_t)i * HW4 + m] = (_Float16)psum[mi][ni][rr];
      }
}

// ---------------------------------------------------------------------------
// out_attn = Asum @ V  -> stored f16 into mem_out low half at [b, i*256+j]
// (raw-view reshape of the [2304,256] product into [256,48,48]).
// B operand: V^T row-major = natural pooled m_val flat [256, 4608].
__global__ __launch_bounds__(128) void k_av(
    const _Float16* __restrict__ asum, const _Float16* __restrict__ mv,
    _Float16* __restrict__ mem) {
  const int lane = threadIdx.x & 31;
  const int tile = blockIdx.x * 4 + (threadIdx.x >> 5);
  const int nt  = tile % 8;
  const int rem = tile / 8;
  const int it  = rem % 72;
  const int b   = rem / 72;
  const int i0 = it * 32, j0 = nt * 32;
  const _Float16* A  = asum + (size_t)b * HW2 * HW4;
  const _Float16* Vt = mv + (size_t)b * VD * HW4;   // V^T [256, 4608]
  const int kofs = (lane >> 4) << 3;
  const int jlo = j0 + (lane & 15), jhi = jlo + 16;

  v8f acc[2][2] = {};
  for (int kb = 0; kb < HW4; kb += 32) {
    v16h a0 = load_row_frag(A + (size_t)(i0 + (lane & 15)) * HW4 + kb + kofs);
    v16h a1 = load_row_frag(A + (size_t)(i0 + 16 + (lane & 15)) * HW4 + kb + kofs);
    v16h b0 = load_row_frag(Vt + (size_t)jlo * HW4 + kb + kofs);
    v16h b1 = load_row_frag(Vt + (size_t)jhi * HW4 + kb + kofs);
    acc[0][0] = wmma16(a0, b0, acc[0][0]);
    acc[0][1] = wmma16(a0, b1, acc[0][1]);
    acc[1][0] = wmma16(a1, b0, acc[1][0]);
    acc[1][1] = wmma16(a1, b1, acc[1][1]);
  }
#pragma unroll
  for (int mi = 0; mi < 2; ++mi)
#pragma unroll
    for (int ni = 0; ni < 2; ++ni)
#pragma unroll
      for (int rr = 0; rr < 8; ++rr) {
        int i = i0 + mi * 16 + rr + ((lane >> 4) << 3);
        int j = j0 + ni * 16 + (lane & 15);
        mem[(size_t)b * 512 * HW2 + (size_t)i * VD + j] = (_Float16)acc[mi][ni][rr];
      }
}

// ---------------------------------------------------------------------------
// nearest-2x-upsample + 3x3 pad-1 conv as implicit GEMM:
// out[co, Y*96+X] = sum_{ci,dy,dx} w[co, ci*9+dy*3+dx] * mem[ci, (Y+dy-1)>>1, (X+dx-1)>>1]
__global__ __launch_bounds__(128) void k_reduce_conv(
    const _Float16* __restrict__ w16, const _Float16* __restrict__ mem,
    float* __restrict__ out) {
  const int lane = threadIdx.x & 31;
  const int tile = blockIdx.x * 4 + (threadIdx.x >> 5);
  const int nt  = tile % 288;
  const int rem = tile / 288;
  const int mt  = rem % 8;
  const int b   = rem / 8;
  const int m0 = mt * 32, n0 = nt * 32;
  const int kofs = (lane >> 4) << 3;
  const _Float16* M = mem + (size_t)b * 512 * HW2;

  int nn[2] = { n0 + (lane & 15), n0 + (lane & 15) + 16 };
  int Y_[2], X_[2];
#pragma unroll
  for (int h = 0; h < 2; ++h) { Y_[h] = nn[h] / 96; X_[h] = nn[h] - Y_[h] * 96; }

  v8f acc[2][2] = {};
  for (int kb = 0; kb < 4608; kb += 32) {
    v16h a0 = load_row_frag(w16 + (size_t)(m0 + (lane & 15)) * 4608 + kb + kofs);
    v16h a1 = load_row_frag(w16 + (size_t)(m0 + 16 + (lane & 15)) * 4608 + kb + kofs);
    v16h bf0, bf1;
#pragma unroll
    for (int e = 0; e < 16; ++e) {
      int kk = kb + kmap(e, lane);
      int ci = kk / 9; int r9 = kk - ci * 9; int dy = r9 / 3; int dx = r9 - dy * 3;
      {
        int yy = Y_[0] + dy - 1, xx = X_[0] + dx - 1;
        bf0[e] = (yy >= 0 && yy < 96 && xx >= 0 && xx < 96)
                     ? M[(size_t)ci * HW2 + (yy >> 1) * 48 + (xx >> 1)] : (_Float16)0.f;
      }
      {
        int yy = Y_[1] + dy - 1, xx = X_[1] + dx - 1;
        bf1[e] = (yy >= 0 && yy < 96 && xx >= 0 && xx < 96)
                     ? M[(size_t)ci * HW2 + (yy >> 1) * 48 + (xx >> 1)] : (_Float16)0.f;
      }
    }
    acc[0][0] = wmma16(a0, bf0, acc[0][0]);
    acc[0][1] = wmma16(a0, bf1, acc[0][1]);
    acc[1][0] = wmma16(a1, bf0, acc[1][0]);
    acc[1][1] = wmma16(a1, bf1, acc[1][1]);
  }
#pragma unroll
  for (int mi = 0; mi < 2; ++mi)
#pragma unroll
    for (int ni = 0; ni < 2; ++ni)
#pragma unroll
      for (int rr = 0; rr < 8; ++rr) {
        int m = m0 + mi * 16 + rr + ((lane >> 4) << 3);
        int n = n0 + ni * 16 + (lane & 15);
        out[(size_t)b * VD * 9216 + (size_t)m * 9216 + n] = acc[mi][ni][rr];
      }
}

// ---------------------------------------------------------------------------
extern "C" void kernel_launch(void* const* d_in, const int* in_sizes, int n_in,
                              void* d_out, int out_size, void* d_ws, size_t ws_size,
                              hipStream_t stream) {
  (void)in_sizes; (void)n_in; (void)out_size; (void)ws_size;

  // ---- workspace layout (halves; total ~124 MB) ----
  _Float16* h    = (_Float16*)d_ws;
  _Float16* wqk  = h;                       // 128*512
  _Float16* wmk  = wqk + 65536;             // 256*1024
  _Float16* wqv  = wmk + 262144;            // 256*1024
  _Float16* wmv  = wqv + 262144;            // 512*2048
  _Float16* wrd  = wmv + 1048576;           // 256*4608
  _Float16* qk   = wrd + 1179648;           // 4*2304*128
  _Float16* pqk  = qk  + 1179648;
  _Float16* mkT  = pqk + 1179648;           // 4*4608*128  (Km^T)
  _Float16* pmkT = mkT + 2359296;
  _Float16* mv   = pmkT + 2359296;          // 4*256*4608  (pooled m_val = V^T)
  _Float16* mem  = mv  + 4718592;           // 4*512*2304  (mem_out, f16)
  _Float16* as_  = mem + 4718592;           // 4*2304*4608 (sum of 4 prob mats)
  float*    cmax = (float*)(as_ + 42467328); // 4*4*4608
  float*    ssum = cmax + 73728;

  // ---- weight f32 -> f16 ----
  k_cvt<<<(65536   + 255) / 256, 256, 0, stream>>>((const float*)d_in[6],  wqk, 65536);
  k_cvt<<<(262144  + 255) / 256, 256, 0, stream>>>((const float*)d_in[7],  wmk, 262144);
  k_cvt<<<(262144  + 255) / 256, 256, 0, stream>>>((const float*)d_in[8],  wqv, 262144);
  k_cvt<<<(1048576 + 255) / 256, 256, 0, stream>>>((const float*)d_in[9],  wmv, 1048576);
  k_cvt<<<(1179648 + 255) / 256, 256, 0, stream>>>((const float*)d_in[10], wrd, 1179648);

  // ---- six 2x2/s2 pooling convs (implicit GEMM, WMMA) ----
  k_pool_conv<<< 288, 128, 0, stream>>>((const float*)d_in[2], wqk, qk,   128, 128, 0); // q_key
  k_pool_conv<<< 288, 128, 0, stream>>>((const float*)d_in[5], wqk, pqk,  128, 128, 0); // p_q_key
  k_pool_conv<<< 576, 128, 0, stream>>>((const float*)d_in[0], wmk, mkT,  256, 256, 1); // m_key   -> Km^T
  k_pool_conv<<< 576, 128, 0, stream>>>((const float*)d_in[4], wmk, pmkT, 256, 256, 1); // p_m_key -> Km^T
  k_pool_conv<<<1152, 128, 0, stream>>>((const float*)d_in[1], wmv, mv,   512, 512, 0); // m_val (flat = V^T)
  k_pool_conv<<< 576, 128, 0, stream>>>((const float*)d_in[3], wqv, mem,  256, 256, 2); // q_val -> mem hi

  // ---- attention: column-softmax stats, summed prob matrix, A@V ----
  k_qk_stats<<<  576, 128, 0, stream>>>(qk, pqk, mkT, pmkT, cmax, ssum);
  k_asum    <<<10368, 128, 0, stream>>>(qk, pqk, mkT, pmkT, cmax, ssum, as_);
  k_av      <<<  576, 128, 0, stream>>>(as_, mv, mem);

  // ---- upsample + 3x3 reduce conv (implicit GEMM, WMMA) -> f32 out ----
  k_reduce_conv<<<2304, 128, 0, stream>>>(wrd, mem, (float*)d_out);
}
